// SelfAttention_7756710936945
// MI455X (gfx1250) — matrix operand order repaired and checked by reading, not compiled, not run on hardware
//
#include <hip/hip_runtime.h>
#include <hip/hip_bf16.h>

// ---------------------------------------------------------------------------
// Self-attention forward for MI455X (gfx1250, wave32, WMMA).
// x:[1,4096,1024] f32, w_qkv:[3072,1024], b_qkv:[3072], w_o:[1024,1024],
// b_o:[1024], causal. Compute in f16 operands / f32 accumulation via
// v_wmma_f32_16x16x32_f16.
// ---------------------------------------------------------------------------

#define SEQ 4096
#define DIM 1024
#define NHEAD 16
#define HDIM 64

typedef __attribute__((ext_vector_type(16))) _Float16 v16h;
typedef __attribute__((ext_vector_type(8)))  _Float16 v8h;
typedef __attribute__((ext_vector_type(8)))  float    v8f;

static __device__ inline v16h pack16(v8h lo, v8h hi) {
    v16h r;
#pragma unroll
    for (int i = 0; i < 8; ++i) { r[i] = lo[i]; r[i + 8] = hi[i]; }
    return r;
}

// A fragment 16x32 (MxK), f16, row-major source with leading dim lda.
// Lanes 0-15: M=lane, K = k0+[0..7] and k0+[16..23]
// Lanes 16-31: M=lane-16, K = k0+[8..15] and k0+[24..31]
static __device__ inline v16h load_a(const _Float16* A, int lda, int m0, int k0, int lane) {
    int m  = m0 + (lane & 15);
    int ks = (lane >> 4) ? 8 : 0;
    const _Float16* p = A + (size_t)m * lda + k0 + ks;
    return pack16(*(const v8h*)p, *(const v8h*)(p + 16));
}

// B fragment 32x16 (KxN) where B[k][n] = W[n][k], W row-major with ld ldb.
// Lanes 0-15: N=lane, K = k0+[0..7] and k0+[16..23];  Lanes 16-31: N=lane-16, K = k0+[8..15] and k0+[24..31]
static __device__ inline v16h load_b(const _Float16* W, int ldb, int n0, int k0, int lane) {
    int n  = n0 + (lane & 15);
    int ko = (lane >> 4) ? 8 : 0;
    const _Float16* p = W + (size_t)n * ldb + k0 + ko;
    return pack16(*(const v8h*)p, *(const v8h*)(p + 16));
}

static __device__ inline v8f wmma_f32(v16h a, v16h b, v8f c) {
    return __builtin_amdgcn_wmma_f32_16x16x32_f16(false, a, false, b, (short)0, c, false, false);
}

// ---------------------------------------------------------------------------
// f32 -> f16 conversion
// ---------------------------------------------------------------------------
__global__ __launch_bounds__(256)
void cvt_f32_f16(const float* __restrict__ src, _Float16* __restrict__ dst, int n) {
    int i = blockIdx.x * blockDim.x + threadIdx.x;
    int stride = gridDim.x * blockDim.x;
    for (; i < n; i += stride) dst[i] = (_Float16)src[i];
}

// ---------------------------------------------------------------------------
// QKV GEMM:  C[m,n] = sum_k Xh[m,k] * Wqkv[n,k] + b[n],  M=4096 N=3072 K=1024
// Epilogue routes: n<1024 -> Q[h][s][d]; <2048 -> K[h][t][d]; else Vt[h][d][t]
// Block: 256 threads = 8 waves (2x4), WG tile 128x256, wave tile 64x64.
// ---------------------------------------------------------------------------
__global__ __launch_bounds__(256)
void qkv_gemm(const _Float16* __restrict__ X, const _Float16* __restrict__ W,
              const float* __restrict__ bias,
              _Float16* __restrict__ Qh, _Float16* __restrict__ Kh,
              _Float16* __restrict__ Vt) {
    const int lane = threadIdx.x & 31;
    const int wave = threadIdx.x >> 5;
    const int m0 = blockIdx.y * 128 + (wave >> 2) * 64;
    const int n0 = blockIdx.x * 256 + (wave & 3) * 64;

    const v8f vz = {0.f, 0.f, 0.f, 0.f, 0.f, 0.f, 0.f, 0.f};
    v8f acc[4][4];
#pragma unroll
    for (int i = 0; i < 4; ++i)
#pragma unroll
        for (int j = 0; j < 4; ++j) acc[i][j] = vz;

    for (int k0 = 0; k0 < DIM; k0 += 32) {
        v16h a[4], b[4];
#pragma unroll
        for (int i = 0; i < 4; ++i) a[i] = load_a(X, DIM, m0 + 16 * i, k0, lane);
#pragma unroll
        for (int j = 0; j < 4; ++j) b[j] = load_b(W, DIM, n0 + 16 * j, k0, lane);
#pragma unroll
        for (int i = 0; i < 4; ++i)
#pragma unroll
            for (int j = 0; j < 4; ++j) acc[i][j] = wmma_f32(a[i], b[j], acc[i][j]);
    }

    const int half = lane >> 4, nl = lane & 15;
#pragma unroll
    for (int j = 0; j < 4; ++j) {
        const int gn = n0 + 16 * j + nl;
        const float bj = bias[gn];
        const int sec = gn >> 10;      // 0=Q 1=K 2=V
        const int c = gn & 1023;
        const int h = c >> 6, d = c & 63;
#pragma unroll
        for (int i = 0; i < 4; ++i) {
#pragma unroll
            for (int v = 0; v < 8; ++v) {
                const int gm = m0 + 16 * i + v + 8 * half;   // sequence position
                const _Float16 hv = (_Float16)(acc[i][j][v] + bj);
                if (sec == 0)      Qh[((size_t)h * SEQ + gm) * HDIM + d] = hv;
                else if (sec == 1) Kh[((size_t)h * SEQ + gm) * HDIM + d] = hv;
                else               Vt[((size_t)h * HDIM + d) * SEQ + gm] = hv;
            }
        }
    }
}

// ---------------------------------------------------------------------------
// Flash attention (causal). grid = (SEQ/128, NHEAD), 8 waves/block,
// each wave owns 16 query rows; key blocks of 32.
// ---------------------------------------------------------------------------
#define NEG_BIG (-1.0e30f)

__global__ __launch_bounds__(256)
void attn_kernel(const _Float16* __restrict__ Qh, const _Float16* __restrict__ Kh,
                 const _Float16* __restrict__ Vt, _Float16* __restrict__ At) {
    const int lane = threadIdx.x & 31;
    const int wave = threadIdx.x >> 5;
    const int h = blockIdx.y;
    const int qrow0 = blockIdx.x * 128 + wave * 16;
    const int half = lane >> 4, nl = lane & 15;

    const _Float16* Qb = Qh + (size_t)h * SEQ * HDIM;
    const _Float16* Kb = Kh + (size_t)h * SEQ * HDIM;
    const _Float16* Vb = Vt + (size_t)h * HDIM * SEQ;

    __shared__ _Float16 ldsP[8][16 * 32];       // 1 KB per wave: P transpose staging
    _Float16* lp = ldsP[wave];

    const v16h aQ0 = load_a(Qb, HDIM, qrow0, 0,  lane);
    const v16h aQ1 = load_a(Qb, HDIM, qrow0, 32, lane);

    const v8f vz = {0.f, 0.f, 0.f, 0.f, 0.f, 0.f, 0.f, 0.f};
    v8f accO[4];
#pragma unroll
    for (int t = 0; t < 4; ++t) accO[t] = vz;
    float m_i[8], l_i[8];
#pragma unroll
    for (int v = 0; v < 8; ++v) { m_i[v] = NEG_BIG; l_i[v] = 0.f; }

    for (int kb = 0; kb <= qrow0 + 15; kb += 32) {
        // ---- scores: S = Q @ K^T (two 16x16 tiles over 32 keys) ----
        v8f s0 = vz, s1 = vz;
        s0 = wmma_f32(aQ0, load_b(Kb, HDIM, kb,      0,  lane), s0);
        s0 = wmma_f32(aQ1, load_b(Kb, HDIM, kb,      32, lane), s0);
        s1 = wmma_f32(aQ0, load_b(Kb, HDIM, kb + 16, 0,  lane), s1);
        s1 = wmma_f32(aQ1, load_b(Kb, HDIM, kb + 16, 32, lane), s1);

        // ---- causal mask (before scale, as in reference) + scale ----
#pragma unroll
        for (int v = 0; v < 8; ++v) {
            const int mg = qrow0 + v + 8 * half;
            s0[v] = (kb + nl      <= mg) ? s0[v] * 0.125f : NEG_BIG;
            s1[v] = (kb + 16 + nl <= mg) ? s1[v] * 0.125f : NEG_BIG;
        }

        // ---- online softmax ----
        float alpha[8];
#pragma unroll
        for (int v = 0; v < 8; ++v) {
            float mx = fmaxf(s0[v], s1[v]);
#pragma unroll
            for (int off = 1; off < 16; off <<= 1)
                mx = fmaxf(mx, __shfl_xor(mx, off, 16));
            const float mnew = fmaxf(m_i[v], mx);
            alpha[v] = __expf(m_i[v] - mnew);
            const float p0 = __expf(s0[v] - mnew);
            const float p1 = __expf(s1[v] - mnew);
            float rs = p0 + p1;
#pragma unroll
            for (int off = 1; off < 16; off <<= 1)
                rs += __shfl_xor(rs, off, 16);
            l_i[v] = l_i[v] * alpha[v] + rs;
            m_i[v] = mnew;
            s0[v] = p0; s1[v] = p1;
        }
#pragma unroll
        for (int t = 0; t < 4; ++t)
#pragma unroll
            for (int v = 0; v < 8; ++v) accO[t][v] *= alpha[v];

        // ---- transpose P (C layout -> A layout) through LDS ----
#pragma unroll
        for (int v = 0; v < 8; ++v) {
            const int row = v + 8 * half;
            lp[row * 32 + nl]      = (_Float16)s0[v];
            lp[row * 32 + 16 + nl] = (_Float16)s1[v];
        }
        asm volatile("s_wait_dscnt 0" ::: "memory");
        const int ks = half ? 8 : 0;
        const _Float16* pp = lp + (lane & 15) * 32 + ks;
        const v16h aP = pack16(*(const v8h*)pp, *(const v8h*)(pp + 16));

        // ---- O += P @ V  (Vt[d][t] makes B-frags contiguous) ----
#pragma unroll
        for (int t = 0; t < 4; ++t)
            accO[t] = wmma_f32(aP, load_b(Vb, SEQ, 16 * t, kb, lane), accO[t]);
    }

    // ---- finalize: divide by l, write head-interleaved f16 [s][h*64+d] ----
#pragma unroll
    for (int t = 0; t < 4; ++t) {
#pragma unroll
        for (int v = 0; v < 8; ++v) {
            const int mg = qrow0 + v + 8 * half;
            const int d = 16 * t + nl;
            At[(size_t)mg * DIM + h * HDIM + d] = (_Float16)(accO[t][v] / l_i[v]);
        }
    }
}

// ---------------------------------------------------------------------------
// Output projection: out[m,n] = sum_k At[m,k]*Wo[n,k] + b_o[n]  (fp32 out)
// ---------------------------------------------------------------------------
__global__ __launch_bounds__(256)
void out_gemm(const _Float16* __restrict__ A, const _Float16* __restrict__ W,
              const float* __restrict__ bias, float* __restrict__ out) {
    const int lane = threadIdx.x & 31;
    const int wave = threadIdx.x >> 5;
    const int m0 = blockIdx.y * 128 + (wave >> 2) * 64;
    const int n0 = blockIdx.x * 256 + (wave & 3) * 64;

    const v8f vz = {0.f, 0.f, 0.f, 0.f, 0.f, 0.f, 0.f, 0.f};
    v8f acc[4][4];
#pragma unroll
    for (int i = 0; i < 4; ++i)
#pragma unroll
        for (int j = 0; j < 4; ++j) acc[i][j] = vz;

    for (int k0 = 0; k0 < DIM; k0 += 32) {
        v16h a[4], b[4];
#pragma unroll
        for (int i = 0; i < 4; ++i) a[i] = load_a(A, DIM, m0 + 16 * i, k0, lane);
#pragma unroll
        for (int j = 0; j < 4; ++j) b[j] = load_b(W, DIM, n0 + 16 * j, k0, lane);
#pragma unroll
        for (int i = 0; i < 4; ++i)
#pragma unroll
            for (int j = 0; j < 4; ++j) acc[i][j] = wmma_f32(a[i], b[j], acc[i][j]);
    }

    const int half = lane >> 4, nl = lane & 15;
#pragma unroll
    for (int j = 0; j < 4; ++j) {
        const int gn = n0 + 16 * j + nl;
        const float bj = bias[gn];
#pragma unroll
        for (int i = 0; i < 4; ++i)
#pragma unroll
            for (int v = 0; v < 8; ++v) {
                const int gm = m0 + 16 * i + v + 8 * half;
                out[(size_t)gm * DIM + gn] = acc[i][j][v] + bj;
            }
    }
}

// ---------------------------------------------------------------------------
extern "C" void kernel_launch(void* const* d_in, const int* in_sizes, int n_in,
                              void* d_out, int out_size, void* d_ws, size_t ws_size,
                              hipStream_t stream) {
    const float* x      = (const float*)d_in[0];
    const float* w_qkv  = (const float*)d_in[1];
    const float* b_qkv  = (const float*)d_in[2];
    const float* w_o    = (const float*)d_in[3];
    const float* b_o    = (const float*)d_in[4];
    (void)n_in; (void)in_sizes; (void)out_size; (void)ws_size;

    char* ws = (char*)d_ws;
    const size_t MB = 1u << 20;
    _Float16* Xh  = (_Float16*)(ws + 0 * MB);   // 8 MiB  [4096][1024]
    _Float16* Wh  = (_Float16*)(ws + 8 * MB);   // 6 MiB  [3072][1024]
    _Float16* WOh = (_Float16*)(ws + 14 * MB);  // 2 MiB  [1024][1024]
    _Float16* Qh  = (_Float16*)(ws + 16 * MB);  // 8 MiB  [16][4096][64]
    _Float16* Kh  = (_Float16*)(ws + 24 * MB);  // 8 MiB  [16][4096][64]
    _Float16* Vt  = (_Float16*)(ws + 32 * MB);  // 8 MiB  [16][64][4096]
    _Float16* At  = (_Float16*)(ws + 40 * MB);  // 8 MiB  [4096][1024]

    cvt_f32_f16<<<2048, 256, 0, stream>>>(x,     Xh,  SEQ * DIM);
    cvt_f32_f16<<<2048, 256, 0, stream>>>(w_qkv, Wh,  3 * DIM * DIM);
    cvt_f32_f16<<<1024, 256, 0, stream>>>(w_o,   WOh, DIM * DIM);

    qkv_gemm<<<dim3(3 * DIM / 256, SEQ / 128), 256, 0, stream>>>(Xh, Wh, b_qkv, Qh, Kh, Vt);
    attn_kernel<<<dim3(SEQ / 128, NHEAD), 256, 0, stream>>>(Qh, Kh, Vt, At);
    out_gemm<<<dim3(DIM / 256, SEQ / 128), 256, 0, stream>>>(At, WOh, b_o, (float*)d_out);
}
